// MaskedCartesianAttention_34952443855040
// MI455X (gfx1250) — compile-verified, matrix operand order checked by reading, not run on hardware
//
#include <hip/hip_runtime.h>

#define BS 4
#define QL 2048
#define SL 2048
#define H  256      // d_model / 2
#define DK 128
#define DH 64
#define NH 4

typedef __bf16 bf16;
typedef __attribute__((ext_vector_type(16))) __bf16 v16bf;
typedef __attribute__((ext_vector_type(8)))  __bf16 v8bf;
typedef __attribute__((ext_vector_type(8)))  float  v8f;
typedef int v4i32 __attribute__((vector_size(16)));   // matches builtin param type

// Native gfx1250 bf16 converts (v_cvt_*_bf16) instead of software rounding.
__device__ __forceinline__ bf16 f2bf(float f) { return (bf16)f; }
__device__ __forceinline__ float bf2f(bf16 v) { return (float)v; }

// A-operand (or B-as-rows-of-B^T) tile load, bf16 row-major source.
// ISA 16-bit A 16x32 layout: lane m=l&15, g=l>>4: elems0-7 = K g*8..g*8+7,
// elems8-15 = K 16+g*8..16+g*8+7.
__device__ __forceinline__ v16bf load_tile_bf16(const bf16* base, int ld, int lane) {
  int m = lane & 15, g = (lane >> 4) & 1;
  v8bf lo = *(const v8bf*)(base + (size_t)m * ld + g * 8);
  v8bf hi = *(const v8bf*)(base + (size_t)m * ld + 16 + g * 8);
  v16bf r;
#pragma unroll
  for (int i = 0; i < 8; i++) { r[i] = lo[i]; r[i + 8] = hi[i]; }
  return r;
}

__device__ __forceinline__ v16bf load_tile_f32(const float* base, int ld, int lane) {
  int m = lane & 15, g = (lane >> 4) & 1;
  const float* p0 = base + (size_t)m * ld + g * 8;
  const float* p1 = base + (size_t)m * ld + 16 + g * 8;
  v16bf r;
#pragma unroll
  for (int i = 0; i < 8; i++) { r[i] = f2bf(p0[i]); r[i + 8] = f2bf(p1[i]); }
  return r;
}

__device__ __forceinline__ v8f wmma_bf16(v16bf a, v16bf b, v8f c) {
  return __builtin_amdgcn_wmma_f32_16x16x32_bf16(false, a, false, b, (short)0, c,
                                                 false, false);
}

// ---- CDNA5 async global->LDS staging (ASYNCcnt path) with portable fallback ----
__device__ __forceinline__ void wait_async() {
#if defined(__gfx1250__)
#if __has_builtin(__builtin_amdgcn_s_wait_asynccnt)
  __builtin_amdgcn_s_wait_asynccnt(0);
#else
  asm volatile("s_wait_asynccnt 0" ::: "memory");
#endif
#endif
}

#if defined(__gfx1250__) && __has_builtin(__builtin_amdgcn_global_load_async_to_lds_b128)
__device__ __forceinline__ void async_ld16B(const bf16* g, bf16* l) {
  __builtin_amdgcn_global_load_async_to_lds_b128(
      (__attribute__((address_space(1))) v4i32*)(v4i32*)(void*)const_cast<bf16*>(g),
      (__attribute__((address_space(3))) v4i32*)(v4i32*)(void*)l, 0, 0);
}
#else
__device__ __forceinline__ void async_ld16B(const bf16* g, bf16* l) {
  *(v8bf*)l = *(const v8bf*)g;
}
#endif

// Stage a 32-token x 128-dim bf16 tile (8KB) row-major into LDS, 256 threads.
// Each thread moves 32 bytes as two async B128 transfers.
__device__ __forceinline__ void stage32x128(bf16* ldsbase, const bf16* gsrc, int t) {
  int ss = t >> 3;             // 0..31 token
  int d0 = (t & 7) * 16;       // 0..112 dim
  const bf16* g = gsrc + (size_t)ss * DK + d0;
  bf16* l = ldsbase + ss * DK + d0;
  async_ld16B(g, l);
  async_ld16B(g + 8, l + 8);
}

// ---------------------------------------------------------------- weights
__global__ void k_transpose_w(const float* __restrict__ W, bf16* __restrict__ WT,
                              int K, int N) {
  int i = blockIdx.x * 256 + threadIdx.x;
  if (i >= K * N) return;
  int n = i / K, k = i % K;
  WT[i] = f2bf(W[(size_t)k * N + n]);
}

// --------------------------------------------------- projection GEMM (K=256,N=128)
__global__ void k_gemm_proj(const float* __restrict__ A, const bf16* __restrict__ WT,
                            bf16* __restrict__ C) {
  const int K = 256, N = 128;
  int lane = threadIdx.x & 31, wave = threadIdx.x >> 5;
  int m0 = blockIdx.x * 128 + wave * 16;
  v16bf a[8];
#pragma unroll
  for (int kc = 0; kc < 8; kc++)
    a[kc] = load_tile_f32(A + (size_t)m0 * K + kc * 32, K, lane);
  int n = lane & 15, g = lane >> 4;
  for (int nt = 0; nt < 8; nt++) {
    v8f acc = {};
#pragma unroll
    for (int kc = 0; kc < 8; kc++) {
      v16bf b = load_tile_bf16(WT + (size_t)(nt * 16) * K + kc * 32, K, lane);
      acc = wmma_bf16(a[kc], b, acc);
    }
#pragma unroll
    for (int r = 0; r < 8; r++)
      C[(size_t)(m0 + r + 8 * g) * N + nt * 16 + n] = f2bf(acc[r]);
  }
}

// ---------------------------------------------- head assembly (a_{h&1}, b_{h>>1})
__global__ void k_build_heads(const bf16* __restrict__ ya, const bf16* __restrict__ yb,
                              bf16* __restrict__ xh, int L) {
  size_t i = (size_t)blockIdx.x * 256 + threadIdx.x;
  size_t total = (size_t)BS * NH * L * DK;
  if (i >= total) return;
  int d = (int)(i & 127);
  size_t t1 = i >> 7;
  int t = (int)(t1 % L); t1 /= L;
  int h = (int)(t1 & 3); int b = (int)(t1 >> 2);
  bf16 v;
  if (d < DH) v = ya[((size_t)(b * L + t)) * 128 + (h & 1) * 64 + d];
  else        v = yb[((size_t)(b * L + t)) * 128 + (h >> 1) * 64 + (d - 64)];
  xh[i] = v;
}

// --------------------------- Q rotation: qh = qh_tmp x E^T per (b,h), WMMA GEMM
__global__ void k_rot_q(const bf16* __restrict__ qh_tmp, const float* __restrict__ E4,
                        bf16* __restrict__ qh) {
  int lane = threadIdx.x & 31, wave = threadIdx.x >> 5;
  int bh = blockIdx.x;
  int b = bh >> 2;
  int m0 = blockIdx.y * 128 + wave * 16;
  const bf16* Abase = qh_tmp + (size_t)bh * QL * DK;
  const float* E = E4 + (size_t)b * DK * DK;
  v16bf a[4];
#pragma unroll
  for (int kc = 0; kc < 4; kc++)
    a[kc] = load_tile_bf16(Abase + (size_t)m0 * DK + kc * 32, DK, lane);
  int n = lane & 15, g = lane >> 4;
  for (int nt = 0; nt < 8; nt++) {
    v8f acc = {};
#pragma unroll
    for (int kc = 0; kc < 4; kc++) {
      v16bf bb = load_tile_f32(E + (size_t)(nt * 16) * DK + kc * 32, DK, lane);
      acc = wmma_bf16(a[kc], bb, acc);
    }
#pragma unroll
    for (int r = 0; r < 8; r++)
      qh[(size_t)bh * QL * DK + (size_t)(m0 + r + 8 * g) * DK + nt * 16 + n] =
          f2bf(acc[r]);
  }
}

// ---------------- K rotation: per-token 128x128 matvec (bound by 537MB embed read)
__global__ void k_rot_k(const bf16* __restrict__ kh_tmp, const float* __restrict__ Ekv,
                        bf16* __restrict__ kh) {
  __shared__ __align__(16) float kv[NH][DK];
  int s = blockIdx.x, b = blockIdx.y;
  int t = threadIdx.x;
  for (int o = t; o < NH * DK; o += 256) {
    int hh = o >> 7, i = o & 127;
    kv[hh][i] = bf2f(kh_tmp[(((size_t)(b * NH + hh) * SL + s) << 7) + i]);
  }
  __syncthreads();
  int i = t & 127;
  int h0 = t >> 7;  // heads h0 and h0+2 share E row i
  const float* Er = Ekv + (((size_t)b * SL + s) * DK + i) * DK;
  float acc0 = 0.f, acc1 = 0.f;
#pragma unroll 4
  for (int j = 0; j < DK; j++) {
    float e = Er[j];
    acc0 += e * kv[h0][j];
    acc1 += e * kv[h0 + 2][j];
  }
  kh[(((size_t)(b * NH + h0) * SL + s) << 7) + i]     = f2bf(acc0);
  kh[(((size_t)(b * NH + h0 + 2) * SL + s) << 7) + i] = f2bf(acc1);
}

// ------------------------------- attention pass A: per-row max & sumexp (flash stats)
// K tile staged once per block via async global->LDS; shared by all 8 waves.
__global__ void k_attn_stats(const bf16* __restrict__ qh, const bf16* __restrict__ kh,
                             const float* __restrict__ bias, const int* __restrict__ mask,
                             float* __restrict__ ms, float* __restrict__ ls) {
  const float SCALE = 0.088388347648318447f;  // 1/sqrt(128)
  __shared__ __align__(16) bf16 KL[32][DK];   // 8KB
  int lane = threadIdx.x & 31, wave = threadIdx.x >> 5;
  int t = threadIdx.x;
  int b = blockIdx.z, h = blockIdx.y;
  int q0 = blockIdx.x * 128 + wave * 16;
  size_t bh = (size_t)(b * NH + h);
  const bf16* Q = qh + (bh * QL + q0) * DK;
  v16bf a[4];
#pragma unroll
  for (int kc = 0; kc < 4; kc++) a[kc] = load_tile_bf16(Q + kc * 32, DK, lane);
  float mrun[8], lrun[8];
#pragma unroll
  for (int r = 0; r < 8; r++) { mrun[r] = -3.0e38f; lrun[r] = 0.f; }
  int n = lane & 15, g = lane >> 4;
  for (int s0 = 0; s0 < SL; s0 += 32) {
    __syncthreads();
    stage32x128(&KL[0][0], kh + (bh * SL + s0) * DK, t);
    wait_async();
    __syncthreads();
#pragma unroll
    for (int st = 0; st < 2; st++) {
      v8f c = {};
#pragma unroll
      for (int kc = 0; kc < 4; kc++) {
        v16bf bb = load_tile_bf16(&KL[st * 16][0] + kc * 32, DK, lane);
        c = wmma_bf16(a[kc], bb, c);
      }
      int sidx = b * SL + s0 + st * 16 + n;
      float addv = -bias[sidx] + (mask[sidx] ? -1.0e9f : 0.f);
#pragma unroll
      for (int r = 0; r < 8; r++) {
        float x = c[r] * SCALE + addv;
        float tm = x;
        tm = fmaxf(tm, __shfl_xor(tm, 1, 32));
        tm = fmaxf(tm, __shfl_xor(tm, 2, 32));
        tm = fmaxf(tm, __shfl_xor(tm, 4, 32));
        tm = fmaxf(tm, __shfl_xor(tm, 8, 32));
        float nm = fmaxf(mrun[r], tm);
        float p = __expf(x - nm);
        p += __shfl_xor(p, 1, 32);
        p += __shfl_xor(p, 2, 32);
        p += __shfl_xor(p, 4, 32);
        p += __shfl_xor(p, 8, 32);
        lrun[r] = lrun[r] * __expf(mrun[r] - nm) + p;
        mrun[r] = nm;
      }
    }
  }
  if (n == 0) {
#pragma unroll
    for (int r = 0; r < 8; r++) {
      size_t idx = bh * QL + q0 + 8 * g + r;
      ms[idx] = mrun[r];
      ls[idx] = lrun[r];
    }
  }
}

// --------- attention pass B: write normalized attn + accumulate P*V (WMMA, LDS)
__global__ void k_attn_out(const bf16* __restrict__ qh, const bf16* __restrict__ kh,
                           const bf16* __restrict__ vh,
                           const float* __restrict__ bias, const int* __restrict__ mask,
                           const float* __restrict__ ms, const float* __restrict__ ls,
                           float* __restrict__ attn, float* __restrict__ oh) {
  const float SCALE = 0.088388347648318447f;
  __shared__ __align__(16) bf16 KL[32][DK];    // K tile, async-staged (8KB)
  __shared__ __align__(16) bf16 VT[DK][32];    // V^T tile: [d][s] (8KB)
  __shared__ __align__(16) bf16 P[8][16][32];  // per-wave prob tile (8KB)
  int lane = threadIdx.x & 31, wave = threadIdx.x >> 5;
  int t = threadIdx.x;
  int b = blockIdx.z, h = blockIdx.y;
  int q0 = blockIdx.x * 128 + wave * 16;
  size_t bh = (size_t)(b * NH + h);
  const bf16* Q = qh + (bh * QL + q0) * DK;
  v16bf a[4];
#pragma unroll
  for (int kc = 0; kc < 4; kc++) a[kc] = load_tile_bf16(Q + kc * 32, DK, lane);
  int n = lane & 15, g = lane >> 4;
  float mrow[8], lrow[8];
#pragma unroll
  for (int r = 0; r < 8; r++) {
    size_t idx = bh * QL + q0 + 8 * g + r;
    mrow[r] = ms[idx];
    float l = ls[idx];
    lrow[r] = (l > 0.f) ? 1.f / l : 0.f;
  }
  v8f acc[8];
  v8f zero = {};
#pragma unroll
  for (int nb = 0; nb < 8; nb++) acc[nb] = zero;

  for (int s0 = 0; s0 < SL; s0 += 32) {
    __syncthreads();
    stage32x128(&KL[0][0], kh + (bh * SL + s0) * DK, t);  // async K
    {  // stage V transposed into LDS (manual: async cannot transpose)
      int ss = t >> 3;
      int d0 = (t & 7) * 16;
      const v8bf* Vr = (const v8bf*)(vh + (bh * SL + s0 + ss) * DK + d0);
      v8bf v0 = Vr[0], v1 = Vr[1];
#pragma unroll
      for (int j = 0; j < 8; j++) { VT[d0 + j][ss] = v0[j]; VT[d0 + 8 + j][ss] = v1[j]; }
    }
    wait_async();
    __syncthreads();
#pragma unroll
    for (int st = 0; st < 2; st++) {
      int sb = s0 + st * 16;
      v8f c = {};
#pragma unroll
      for (int kc = 0; kc < 4; kc++) {
        v16bf bb = load_tile_bf16(&KL[st * 16][0] + kc * 32, DK, lane);
        c = wmma_bf16(a[kc], bb, c);
      }
      int sidx = b * SL + sb + n;
      float addv = -bias[sidx] + (mask[sidx] ? -1.0e9f : 0.f);
#pragma unroll
      for (int r = 0; r < 8; r++) {
        float p = __expf(c[r] * SCALE + addv - mrow[r]) * lrow[r];
        attn[(bh * QL + q0 + 8 * g + r) * SL + sb + n] = p;
        P[wave][8 * g + r][st * 16 + n] = f2bf(p);
      }
    }
    __syncthreads();
    v16bf ap = load_tile_bf16(&P[wave][0][0], 32, lane);
#pragma unroll
    for (int nb = 0; nb < 8; nb++) {
      v16bf bb = load_tile_bf16(&VT[nb * 16][0], 32, lane);
      acc[nb] = wmma_bf16(ap, bb, acc[nb]);
    }
  }
#pragma unroll
  for (int nb = 0; nb < 8; nb++)
#pragma unroll
    for (int r = 0; r < 8; r++)
      oh[(bh * QL + q0 + 8 * g + r) * DK + nb * 16 + n] = acc[nb][r];
}

// ------------------- output projection with head-interleave gather folded into A load
__global__ void k_out_proj(const float* __restrict__ oh, const bf16* __restrict__ WToa,
                           const bf16* __restrict__ WTob, float* __restrict__ outp) {
  int lane = threadIdx.x & 31, wave = threadIdx.x >> 5;
  int half = blockIdx.y;
  const bf16* WT = half ? WTob : WToa;
  int m0 = blockIdx.x * 128 + wave * 16;  // over BS*QL
  int b = m0 / QL;
  int q0 = m0 % QL;
  v16bf a[8];
#pragma unroll
  for (int kc = 0; kc < 8; kc++) {
    int hh = kc >> 1;
    int dbase = (kc & 1) * 32 + half * 64;
    const float* Ab = oh + ((size_t)(b * NH + hh) * QL + q0) * DK + dbase;
    a[kc] = load_tile_f32(Ab, DK, lane);
  }
  int n = lane & 15, g = lane >> 4;
  for (int nt = 0; nt < 16; nt++) {
    v8f acc = {};
#pragma unroll
    for (int kc = 0; kc < 8; kc++) {
      v16bf bb = load_tile_bf16(WT + (size_t)(nt * 16) * 256 + kc * 32, 256, lane);
      acc = wmma_bf16(a[kc], bb, acc);
    }
#pragma unroll
    for (int r = 0; r < 8; r++) {
      size_t row = (size_t)m0 + 8 * g + r;
      outp[((size_t)half * BS * QL + row) * 256 + nt * 16 + n] = acc[r];
    }
  }
}

extern "C" void kernel_launch(void* const* d_in, const int* in_sizes, int n_in,
                              void* d_out, int out_size, void* d_ws, size_t ws_size,
                              hipStream_t stream) {
  (void)in_sizes; (void)n_in; (void)out_size; (void)ws_size;
  const float* q    = (const float*)d_in[0];   // [2,BS,QL,H]
  const float* k    = (const float*)d_in[1];
  const float* v    = (const float*)d_in[2];
  const float* qse  = (const float*)d_in[3];   // [BS,1,DK,DK]
  const float* kvse = (const float*)d_in[4];   // [BS,SL,DK,DK]
  const float* bias = (const float*)d_in[5];
  const int*   mask = (const int*)d_in[6];
  const float* Wqa = (const float*)d_in[7];
  const float* Wqb = (const float*)d_in[8];
  const float* Wka = (const float*)d_in[9];
  const float* Wkb = (const float*)d_in[10];
  const float* Wva = (const float*)d_in[11];
  const float* Wvb = (const float*)d_in[12];
  const float* Woa = (const float*)d_in[13];
  const float* Wob = (const float*)d_in[14];

  const int M = BS * QL;  // 8192
  char* ws = (char*)d_ws;
  size_t off = 0;
  auto nxt = [&](size_t bytes) {
    void* p = (void*)(ws + off);
    off += (bytes + 255) & ~(size_t)255;
    return p;
  };
  bf16* wtqa = (bf16*)nxt((size_t)128 * 256 * 2);
  bf16* wtqb = (bf16*)nxt((size_t)128 * 256 * 2);
  bf16* wtka = (bf16*)nxt((size_t)128 * 256 * 2);
  bf16* wtkb = (bf16*)nxt((size_t)128 * 256 * 2);
  bf16* wtva = (bf16*)nxt((size_t)128 * 256 * 2);
  bf16* wtvb = (bf16*)nxt((size_t)128 * 256 * 2);
  bf16* wtoa = (bf16*)nxt((size_t)256 * 256 * 2);
  bf16* wtob = (bf16*)nxt((size_t)256 * 256 * 2);
  bf16* yqa = (bf16*)nxt((size_t)M * 128 * 2);
  bf16* yqb = (bf16*)nxt((size_t)M * 128 * 2);
  bf16* yka = (bf16*)nxt((size_t)M * 128 * 2);
  bf16* ykb = (bf16*)nxt((size_t)M * 128 * 2);
  bf16* yva = (bf16*)nxt((size_t)M * 128 * 2);
  bf16* yvb = (bf16*)nxt((size_t)M * 128 * 2);
  size_t hbytes = (size_t)BS * NH * QL * DK * 2;
  bf16* qh_tmp = (bf16*)nxt(hbytes);
  bf16* kh_tmp = (bf16*)nxt(hbytes);
  bf16* qh = (bf16*)nxt(hbytes);
  bf16* kh = (bf16*)nxt(hbytes);
  bf16* vh = (bf16*)nxt(hbytes);
  float* ms = (float*)nxt((size_t)BS * NH * QL * 4);
  float* ls = (float*)nxt((size_t)BS * NH * QL * 4);
  float* ohf = (float*)nxt((size_t)BS * NH * QL * DK * 4);

  dim3 blk(256);
  k_transpose_w<<<dim3((256 * 128 + 255) / 256), blk, 0, stream>>>(Wqa, wtqa, 256, 128);
  k_transpose_w<<<dim3((256 * 128 + 255) / 256), blk, 0, stream>>>(Wqb, wtqb, 256, 128);
  k_transpose_w<<<dim3((256 * 128 + 255) / 256), blk, 0, stream>>>(Wka, wtka, 256, 128);
  k_transpose_w<<<dim3((256 * 128 + 255) / 256), blk, 0, stream>>>(Wkb, wtkb, 256, 128);
  k_transpose_w<<<dim3((256 * 128 + 255) / 256), blk, 0, stream>>>(Wva, wtva, 256, 128);
  k_transpose_w<<<dim3((256 * 128 + 255) / 256), blk, 0, stream>>>(Wvb, wtvb, 256, 128);
  k_transpose_w<<<dim3((256 * 256 + 255) / 256), blk, 0, stream>>>(Woa, wtoa, 256, 256);
  k_transpose_w<<<dim3((256 * 256 + 255) / 256), blk, 0, stream>>>(Wob, wtob, 256, 256);

  const size_t half_in = (size_t)M * H;  // offset of q[1]/k[1]/v[1]
  k_gemm_proj<<<dim3(M / 128), blk, 0, stream>>>(q, wtqa, yqa);
  k_gemm_proj<<<dim3(M / 128), blk, 0, stream>>>(q + half_in, wtqb, yqb);
  k_gemm_proj<<<dim3(M / 128), blk, 0, stream>>>(k, wtka, yka);
  k_gemm_proj<<<dim3(M / 128), blk, 0, stream>>>(k + half_in, wtkb, ykb);
  k_gemm_proj<<<dim3(M / 128), blk, 0, stream>>>(v, wtva, yva);
  k_gemm_proj<<<dim3(M / 128), blk, 0, stream>>>(v + half_in, wtvb, yvb);

  int hb_blocks = (BS * NH * QL * DK) / 256;
  k_build_heads<<<dim3(hb_blocks), blk, 0, stream>>>(yqa, yqb, qh_tmp, QL);
  k_build_heads<<<dim3(hb_blocks), blk, 0, stream>>>(yka, ykb, kh_tmp, SL);
  k_build_heads<<<dim3(hb_blocks), blk, 0, stream>>>(yva, yvb, vh, SL);

  k_rot_q<<<dim3(BS * NH, QL / 128), blk, 0, stream>>>(qh_tmp, qse, qh);
  k_rot_k<<<dim3(SL, BS), blk, 0, stream>>>(kh_tmp, kvse, kh);

  k_attn_stats<<<dim3(QL / 128, NH, BS), blk, 0, stream>>>(qh, kh, bias, mask, ms, ls);

  float* attn = (float*)d_out + (size_t)2 * BS * QL * H;
  k_attn_out<<<dim3(QL / 128, NH, BS), blk, 0, stream>>>(qh, kh, vh, bias, mask, ms, ls,
                                                         attn, ohf);
  k_out_proj<<<dim3(M / 128, 2), blk, 0, stream>>>(ohf, wtoa, wtob, (float*)d_out);
}
